// TransformerBlock_19748259627060
// MI455X (gfx1250) — matrix operand order repaired and checked by reading, not compiled, not run on hardware
//
#include <hip/hip_runtime.h>
#include <hip/hip_bf16.h>

typedef __attribute__((ext_vector_type(16))) _Float16 v16h;
typedef __attribute__((ext_vector_type(8)))  float    v8f;

#define EMB    1024
#define SEQ    2048
#define BATCH  2
#define FFDIM  4096
#define ROWS   (BATCH * SEQ)   /* 4096 token rows */
#define NHEADS 16
#define HDIM   64

#if defined(__AMDGCN__) && __has_builtin(__builtin_amdgcn_global_load_async_to_lds_b128)
#define ASYNC_OK 1
#else
#define ASYNC_OK 0
#endif

// 16-byte copy global -> LDS; async engine (ASYNCcnt) when available.
// Builtin signature (from hipcc diagnostic): param0 = int32x4 addrspace(1)*,
// so cast through void* to get proper addrspacecasts.
static __device__ __forceinline__ void cp16_g2l(_Float16* dst_lds,
                                                const _Float16* src_g) {
#if ASYNC_OK
  typedef int i32x4_v __attribute__((vector_size(16)));
  __builtin_amdgcn_global_load_async_to_lds_b128(
      (__attribute__((address_space(1))) i32x4_v*)(void*)src_g,
      (__attribute__((address_space(3))) i32x4_v*)(void*)dst_lds,
      0, 0);
#else
  *reinterpret_cast<uint4*>(dst_lds) = *reinterpret_cast<const uint4*>(src_g);
#endif
}

static __device__ __forceinline__ void cp_async_wait() {
#if ASYNC_OK
#if __has_builtin(__builtin_amdgcn_s_wait_asynccnt)
  __builtin_amdgcn_s_wait_asynccnt(0);
#else
  asm volatile("s_wait_asynccnt 0x0" ::: "memory");
#endif
#endif
}

static __device__ __forceinline__ v8f wmma_f16(v16h a, v16h b, v8f c) {
  return __builtin_amdgcn_wmma_f32_16x16x32_f16(false, a, false, b, (short)0, c,
                                                false, false);
}

// A-fragment (16x32 f16): lanes 0-15 hold M=0..15 with K=0..15; lanes 16-31
// hold K=16..31 (element e gives K=16*half+e, matching the B fragments).
static __device__ __forceinline__ v16h load_a_frag(const _Float16* p, int stride) {
  const int l = threadIdx.x & 31;
  const int m = l & 15, hh = l >> 4;
  v16h a;
#pragma unroll
  for (int e = 0; e < 16; ++e) {
    const int kk = hh * 16 + e;
    a[e] = p[m * stride + kk];
  }
  return a;
}

// B-fragment (32x16 f16): lane = N (per 16-lane half), element e gives K=16*half+e.
static __device__ __forceinline__ v16h load_b_frag(const _Float16* p, int stride) {
  const int l = threadIdx.x & 31;
  const int n = l & 15, hh = l >> 4;
  v16h b;
#pragma unroll
  for (int e = 0; e < 16; ++e)
    b[e] = p[(hh * 16 + e) * stride + n];
  return b;
}

// Transposed B: logical B[k][n] = p[n*stride + k]  (used for Q*K^T).
static __device__ __forceinline__ v16h load_bt_frag(const _Float16* p, int stride) {
  const int l = threadIdx.x & 31;
  const int n = l & 15, hh = l >> 4;
  v16h b;
#pragma unroll
  for (int e = 0; e < 16; ++e)
    b[e] = p[n * stride + hh * 16 + e];
  return b;
}

static __device__ __forceinline__ float gelu_tanh_f(float x) {
  const float c = 0.7978845608028654f; // sqrt(2/pi)
  return 0.5f * x * (1.0f + tanhf(c * (x + 0.044715f * x * x * x)));
}

// ---------------------------------------------------------------- weight cast
__global__ void cast_f32_to_f16(const float* __restrict__ in,
                                _Float16* __restrict__ out, int n) {
  int i = blockIdx.x * 256 + threadIdx.x;
  if (i < n) out[i] = (_Float16)in[i];
}

// ---------------------------------------------------------------- layernorm
__global__ __launch_bounds__(256)
void layernorm_to_f16(const float* __restrict__ x, const float* __restrict__ gamma,
                      const float* __restrict__ beta, _Float16* __restrict__ out) {
  const int row = blockIdx.x;
  const float* xr = x + (size_t)row * EMB;
  float s = 0.f, ss = 0.f;
  for (int i = threadIdx.x; i < EMB; i += 256) {
    float v = xr[i];
    s += v; ss += v * v;
  }
#pragma unroll
  for (int off = 16; off >= 1; off >>= 1) {
    s  += __shfl_xor(s,  off);
    ss += __shfl_xor(ss, off);
  }
  __shared__ float red[2][8];
  const int wave = threadIdx.x >> 5;
  if ((threadIdx.x & 31) == 0) { red[0][wave] = s; red[1][wave] = ss; }
  __syncthreads();
  s = 0.f; ss = 0.f;
#pragma unroll
  for (int w = 0; w < 8; ++w) { s += red[0][w]; ss += red[1][w]; }
  const float mean = s * (1.0f / EMB);
  const float var  = ss * (1.0f / EMB) - mean * mean;
  const float inv  = rsqrtf(var + 1e-5f);
  _Float16* orow = out + (size_t)row * EMB;
  for (int i = threadIdx.x; i < EMB; i += 256)
    orow[i] = (_Float16)((xr[i] - mean) * inv * gamma[i] + beta[i]);
}

// ---------------------------------------------------------------- WMMA GEMM
// C[M,N] = epilogue(A[M,K] @ B[K,N]); block tile 128x128, 8 waves of 32x64.
// Async double-buffered K staging: copy stage s+1 HBM->LDS while WMMAs chew
// stage s; s_wait_asynccnt + barrier closes each stage.
template <bool BIAS, bool GELU, bool RES, bool OUT16>
__global__ __launch_bounds__(256)
void gemm_f16_wmma(const _Float16* __restrict__ A, const _Float16* __restrict__ Bm,
                   const float* __restrict__ bias, const float* __restrict__ res,
                   void* __restrict__ outp, int M, int N, int K) {
  (void)M;
  __shared__ _Float16 As[2][128 * 64];   // 2 x 16 KB
  __shared__ _Float16 Bs[2][64 * 128];   // 2 x 16 KB
  const int tid  = threadIdx.x;
  const int bm   = blockIdx.y * 128;
  const int bn   = blockIdx.x * 128;
  const int wave = tid >> 5;
  const int wm   = (wave & 3) * 32;
  const int wn   = (wave >> 2) * 64;
  const int lane = tid & 31;
  const int n16  = lane & 15, hh = lane >> 4;

  v8f acc[2][4];
#pragma unroll
  for (int i = 0; i < 2; ++i)
#pragma unroll
    for (int j = 0; j < 4; ++j)
#pragma unroll
      for (int r = 0; r < 8; ++r) acc[i][j][r] = 0.0f;

  auto stage = [&](int buf, int k0) {
    for (int i = tid; i < 1024; i += 256) {             // 128x64 halves / 8
      int r = i >> 3, c = (i & 7) << 3;
      cp16_g2l(&As[buf][r * 64 + c], &A[(size_t)(bm + r) * K + k0 + c]);
    }
    for (int i = tid; i < 1024; i += 256) {             // 64x128 halves / 8
      int r = i >> 4, c = (i & 15) << 3;
      cp16_g2l(&Bs[buf][r * 128 + c], &Bm[(size_t)(k0 + r) * N + bn + c]);
    }
  };

  stage(0, 0);
  cp_async_wait();
  __syncthreads();

  const int nstage = K / 64;
  for (int s = 0; s < nstage; ++s) {
    const int cur = s & 1;
    if (s + 1 < nstage) stage(cur ^ 1, (s + 1) * 64);
#pragma unroll
    for (int ks = 0; ks < 64; ks += 32) {
      v16h a0 = load_a_frag(&As[cur][(wm + 0)  * 64 + ks], 64);
      v16h a1 = load_a_frag(&As[cur][(wm + 16) * 64 + ks], 64);
#pragma unroll
      for (int tj = 0; tj < 4; ++tj) {
        v16h b = load_b_frag(&Bs[cur][ks * 128 + wn + tj * 16], 128);
        acc[0][tj] = wmma_f16(a0, b, acc[0][tj]);
        acc[1][tj] = wmma_f16(a1, b, acc[1][tj]);
      }
    }
    cp_async_wait();
    __syncthreads();
  }

#pragma unroll
  for (int ti = 0; ti < 2; ++ti)
#pragma unroll
    for (int tj = 0; tj < 4; ++tj) {
      const int col = bn + wn + tj * 16 + n16;
      const float bv = BIAS ? bias[col] : 0.0f;
#pragma unroll
      for (int r = 0; r < 8; ++r) {
        const int row = bm + wm + ti * 16 + r + 8 * hh;
        float v = acc[ti][tj][r] + bv;
        if (GELU) v = gelu_tanh_f(v);
        if (RES)  v += res[(size_t)row * N + col];
        if (OUT16) ((_Float16*)outp)[(size_t)row * N + col] = (_Float16)v;
        else       ((float*)outp)[(size_t)row * N + col] = v;
      }
    }
}

// ---------------------------------------------------------------- attention
// One block = (batch, head, 128 q rows); 4 waves, each owns a 32-row q tile.
__global__ __launch_bounds__(128)
void attn_flash(const _Float16* __restrict__ q, const _Float16* __restrict__ kmat,
                const _Float16* __restrict__ vmat, _Float16* __restrict__ ctx) {
  __shared__ _Float16 Kt[32 * 64];
  __shared__ _Float16 Vt[32 * 64];
  __shared__ _Float16 Pbuf[4][32 * 32];
  const int tid  = threadIdx.x;
  const int wave = tid >> 5;
  const int lane = tid & 31;
  const int n16  = lane & 15, hh = lane >> 4;
  const int b = blockIdx.z, head = blockIdx.y, qb = blockIdx.x;
  const int q0 = qb * 128 + wave * 32;
  const size_t rowbase = (size_t)b * SEQ;
  const size_t cbase   = (size_t)head * HDIM;

  // Q fragments live in registers for the whole pass
  v16h aq[2][2];
#pragma unroll
  for (int mi = 0; mi < 2; ++mi)
#pragma unroll
    for (int kb = 0; kb < 2; ++kb)
      aq[mi][kb] =
          load_a_frag(q + (rowbase + q0 + mi * 16) * EMB + cbase + kb * 32, EMB);

  v8f acc[2][4];
#pragma unroll
  for (int mi = 0; mi < 2; ++mi)
#pragma unroll
    for (int nj = 0; nj < 4; ++nj)
#pragma unroll
      for (int r = 0; r < 8; ++r) acc[mi][nj][r] = 0.0f;

  float mrow[2][8], lrow[2][8];
#pragma unroll
  for (int mi = 0; mi < 2; ++mi)
#pragma unroll
    for (int r = 0; r < 8; ++r) { mrow[mi][r] = -__builtin_inff(); lrow[mi][r] = 0.f; }

  const int ktiles = qb * 4 + 4; // causal: only tiles with k <= last q row
  const float scale = 0.125f;    // 1/sqrt(64)

  for (int kt = 0; kt < ktiles; ++kt) {
    __syncthreads();
    for (int i = tid; i < 256; i += 128) {              // 32x64 halves / 8
      int r = i >> 3, c = (i & 7) << 3;
      cp16_g2l(&Kt[r * 64 + c], &kmat[(rowbase + kt * 32 + r) * EMB + cbase + c]);
      cp16_g2l(&Vt[r * 64 + c], &vmat[(rowbase + kt * 32 + r) * EMB + cbase + c]);
    }
    cp_async_wait();
    __syncthreads();
    if (kt * 32 > q0 + 31) continue;   // fully masked for this wave's rows

    // scores = Q (32x64) @ K^T (64x32)
    v8f sc[2][2];
#pragma unroll
    for (int mi = 0; mi < 2; ++mi)
#pragma unroll
      for (int ni = 0; ni < 2; ++ni)
#pragma unroll
        for (int r = 0; r < 8; ++r) sc[mi][ni][r] = 0.0f;
#pragma unroll
    for (int ni = 0; ni < 2; ++ni)
#pragma unroll
      for (int kb = 0; kb < 2; ++kb) {
        v16h bt = load_bt_frag(&Kt[(ni * 16) * 64 + kb * 32], 64);
        sc[0][ni] = wmma_f16(aq[0][kb], bt, sc[0][ni]);
        sc[1][ni] = wmma_f16(aq[1][kb], bt, sc[1][ni]);
      }

    // scale + causal mask
#pragma unroll
    for (int mi = 0; mi < 2; ++mi)
#pragma unroll
      for (int r = 0; r < 8; ++r) {
        const int row = q0 + mi * 16 + r + 8 * hh;
#pragma unroll
        for (int ni = 0; ni < 2; ++ni) {
          const int col = kt * 32 + ni * 16 + n16;
          float s = sc[mi][ni][r] * scale;
          sc[mi][ni][r] = (col > row) ? -__builtin_inff() : s;
        }
      }

    // online softmax (rows live in 16-lane halves of the C layout)
#pragma unroll
    for (int mi = 0; mi < 2; ++mi)
#pragma unroll
      for (int r = 0; r < 8; ++r) {
        float tm = fmaxf(sc[mi][0][r], sc[mi][1][r]);
#pragma unroll
        for (int off = 8; off >= 1; off >>= 1) tm = fmaxf(tm, __shfl_xor(tm, off));
        const float mold = mrow[mi][r];
        const float mnew = fmaxf(mold, tm);
        const float corr = __expf(mold - mnew);
        const float p0 = __expf(sc[mi][0][r] - mnew);
        const float p1 = __expf(sc[mi][1][r] - mnew);
        float ps = p0 + p1;
#pragma unroll
        for (int off = 8; off >= 1; off >>= 1) ps += __shfl_xor(ps, off);
        lrow[mi][r] = lrow[mi][r] * corr + ps;
        mrow[mi][r] = mnew;
        const int prow = mi * 16 + r + 8 * hh;
        Pbuf[wave][prow * 32 + n16]      = (_Float16)p0;
        Pbuf[wave][prow * 32 + 16 + n16] = (_Float16)p1;
#pragma unroll
        for (int nj = 0; nj < 4; ++nj) acc[mi][nj][r] *= corr;
      }

    // ctx += P (32x32) @ V (32x64)  — P re-read from LDS in A layout
    v16h ap0 = load_a_frag(&Pbuf[wave][0],       32);
    v16h ap1 = load_a_frag(&Pbuf[wave][16 * 32], 32);
#pragma unroll
    for (int nj = 0; nj < 4; ++nj) {
      v16h bv = load_b_frag(&Vt[nj * 16], 64);
      acc[0][nj] = wmma_f16(ap0, bv, acc[0][nj]);
      acc[1][nj] = wmma_f16(ap1, bv, acc[1][nj]);
    }
  }

  // normalize + store f16 ctx in [B,S,D] layout (head block of columns)
#pragma unroll
  for (int mi = 0; mi < 2; ++mi)
#pragma unroll
    for (int r = 0; r < 8; ++r) {
      const float inv = 1.0f / lrow[mi][r];
      const int row = q0 + mi * 16 + r + 8 * hh;
#pragma unroll
      for (int nj = 0; nj < 4; ++nj)
        ctx[(rowbase + row) * EMB + cbase + nj * 16 + n16] =
            (_Float16)(acc[mi][nj][r] * inv);
    }
}

// ---------------------------------------------------------------- launcher
extern "C" void kernel_launch(void* const* d_in, const int* in_sizes, int n_in,
                              void* d_out, int out_size, void* d_ws, size_t ws_size,
                              hipStream_t stream) {
  (void)in_sizes; (void)n_in; (void)out_size; (void)ws_size;
  const float* x    = (const float*)d_in[0];
  const float* Wq   = (const float*)d_in[1];
  const float* Wk   = (const float*)d_in[2];
  const float* Wv   = (const float*)d_in[3];
  const float* Wo   = (const float*)d_in[4];
  const float* bo   = (const float*)d_in[5];
  const float* W1   = (const float*)d_in[6];
  const float* b1   = (const float*)d_in[7];
  const float* W2   = (const float*)d_in[8];
  const float* b2   = (const float*)d_in[9];
  const float* ln1s = (const float*)d_in[10];
  const float* ln1b = (const float*)d_in[11];
  const float* ln2s = (const float*)d_in[12];
  const float* ln2b = (const float*)d_in[13];

  char* ws = (char*)d_ws;
  size_t off = 0;
  auto alloc = [&](size_t bytes) -> void* {
    void* p = ws + off;
    off += (bytes + 255) & ~(size_t)255;
    return p;
  };
  _Float16* Wq16  = (_Float16*)alloc((size_t)EMB * EMB * 2);
  _Float16* Wk16  = (_Float16*)alloc((size_t)EMB * EMB * 2);
  _Float16* Wv16  = (_Float16*)alloc((size_t)EMB * EMB * 2);
  _Float16* Wo16  = (_Float16*)alloc((size_t)EMB * EMB * 2);
  _Float16* W116  = (_Float16*)alloc((size_t)EMB * FFDIM * 2);
  _Float16* W216  = (_Float16*)alloc((size_t)FFDIM * EMB * 2);
  _Float16* h16   = (_Float16*)alloc((size_t)ROWS * EMB * 2);
  _Float16* q16   = (_Float16*)alloc((size_t)ROWS * EMB * 2);
  _Float16* k16   = (_Float16*)alloc((size_t)ROWS * EMB * 2);
  _Float16* v16   = (_Float16*)alloc((size_t)ROWS * EMB * 2);
  _Float16* ctx16 = (_Float16*)alloc((size_t)ROWS * EMB * 2);
  float*    x1    = (float*)   alloc((size_t)ROWS * EMB * 4);
  _Float16* h2_16 = (_Float16*)alloc((size_t)ROWS * EMB * 2);
  _Float16* act16 = (_Float16*)alloc((size_t)ROWS * FFDIM * 2);

  const int nw = EMB * EMB, nf = EMB * FFDIM;
  cast_f32_to_f16<<<(nw + 255) / 256, 256, 0, stream>>>(Wq, Wq16, nw);
  cast_f32_to_f16<<<(nw + 255) / 256, 256, 0, stream>>>(Wk, Wk16, nw);
  cast_f32_to_f16<<<(nw + 255) / 256, 256, 0, stream>>>(Wv, Wv16, nw);
  cast_f32_to_f16<<<(nw + 255) / 256, 256, 0, stream>>>(Wo, Wo16, nw);
  cast_f32_to_f16<<<(nf + 255) / 256, 256, 0, stream>>>(W1, W116, nf);
  cast_f32_to_f16<<<(nf + 255) / 256, 256, 0, stream>>>(W2, W216, nf);

  layernorm_to_f16<<<ROWS, 256, 0, stream>>>(x, ln1s, ln1b, h16);

  dim3 g1(EMB / 128, ROWS / 128);    // 8 x 32
  gemm_f16_wmma<false, false, false, true><<<g1, 256, 0, stream>>>(
      h16, Wq16, nullptr, nullptr, q16, ROWS, EMB, EMB);
  gemm_f16_wmma<false, false, false, true><<<g1, 256, 0, stream>>>(
      h16, Wk16, nullptr, nullptr, k16, ROWS, EMB, EMB);
  gemm_f16_wmma<false, false, false, true><<<g1, 256, 0, stream>>>(
      h16, Wv16, nullptr, nullptr, v16, ROWS, EMB, EMB);

  dim3 ga(SEQ / 128, NHEADS, BATCH); // 16 x 16 x 2
  attn_flash<<<ga, 128, 0, stream>>>(q16, k16, v16, ctx16);

  gemm_f16_wmma<true, false, true, false><<<g1, 256, 0, stream>>>(
      ctx16, Wo16, bo, x, x1, ROWS, EMB, EMB);

  layernorm_to_f16<<<ROWS, 256, 0, stream>>>(x1, ln2s, ln2b, h2_16);

  dim3 g2(FFDIM / 128, ROWS / 128); // 32 x 32
  gemm_f16_wmma<true, true, false, true><<<g2, 256, 0, stream>>>(
      h2_16, W116, b1, nullptr, act16, ROWS, FFDIM, EMB);

  gemm_f16_wmma<true, false, true, false><<<g1, 256, 0, stream>>>(
      act16, W216, b2, x1, d_out, ROWS, EMB, FFDIM);
}
